// MultiHeadGraphAttention_91122026152130
// MI455X (gfx1250) — matrix operand order repaired and checked by reading, not compiled, not run on hardware
//
#include <hip/hip_runtime.h>
#include <hip/hip_bf16.h>

#define N_HEAD 4
#define NN     6144
#define F_IN   64
#define F_OUT  32
#define NEG_SLOPE 0.2f

typedef __attribute__((ext_vector_type(16))) __bf16 v16bf;
typedef __attribute__((ext_vector_type(8)))  float  v8f;
typedef __attribute__((ext_vector_type(4)))  float  v4f;

__device__ __forceinline__ float lrelu(float x) { return x >= 0.f ? x : NEG_SLOPE * x; }

// ---------------------------------------------------------------------------
// Kernel 1: h_prime = h @ w per head; store bf16 transposed [head][o][j] for
// WMMA B operand; also reduce attn_src / attn_dst scores per node.
// ---------------------------------------------------------------------------
__global__ void gat_proj_kernel(const float* __restrict__ h,
                                const float* __restrict__ w,
                                const float* __restrict__ a_src,
                                const float* __restrict__ a_dst,
                                __bf16* __restrict__ hpT,
                                float* __restrict__ attn_src,
                                float* __restrict__ attn_dst) {
    __shared__ float sh[8 * F_IN];
    const int head = blockIdx.y;
    const int n0   = blockIdx.x * 8;
    const int tid  = threadIdx.x;
    for (int idx = tid; idx < 8 * F_IN; idx += 256)
        sh[idx] = h[(size_t)(n0 + (idx >> 6)) * F_IN + (idx & 63)];
    __syncthreads();

    const int o = tid & 31;          // output feature (one wave == one node row)
    const int r = tid >> 5;          // local node row
    const int n = n0 + r;
    const float* wh = w + (size_t)head * F_IN * F_OUT + o;
    float hp = 0.f;
#pragma unroll
    for (int f = 0; f < F_IN; ++f)
        hp = fmaf(sh[r * F_IN + f], wh[(size_t)f * F_OUT], hp);

    hpT[((size_t)head * F_OUT + o) * NN + n] = (__bf16)hp;

    float s = hp * a_src[head * F_OUT + o];
    float d = hp * a_dst[head * F_OUT + o];
#pragma unroll
    for (int off = 16; off > 0; off >>= 1) {
        s += __shfl_xor(s, off, 32);
        d += __shfl_xor(d, off, 32);
    }
    if (o == 0) {
        attn_src[(size_t)head * NN + n] = s;
        attn_dst[(size_t)head * NN + n] = d;
    }
}

// ---------------------------------------------------------------------------
// Kernel 2: per-head max over attn_dst (softmax max folds to a closed form:
// m_i = lrelu(s_i + max_j d_j) because leaky-relu is monotonic).
// ---------------------------------------------------------------------------
__global__ void gat_dmax_kernel(const float* __restrict__ attn_dst,
                                float* __restrict__ dmax) {
    __shared__ float red[8];
    const int head = blockIdx.x;
    const int tid  = threadIdx.x;
    float mx = -3.4e38f;
    for (int j = tid; j < NN; j += 256)
        mx = fmaxf(mx, attn_dst[(size_t)head * NN + j]);
#pragma unroll
    for (int off = 16; off > 0; off >>= 1)
        mx = fmaxf(mx, __shfl_xor(mx, off, 32));
    if ((tid & 31) == 0) red[tid >> 5] = mx;
    __syncthreads();
    if (tid == 0) {
        float m = red[0];
#pragma unroll
        for (int i = 1; i < 8; ++i) m = fmaxf(m, red[i]);
        dmax[head] = m;
    }
}

// ---------------------------------------------------------------------------
// Kernel 3: fused softmax + attention output. One wave per 16-row tile.
// Pass 1 over j: denominators (exp only, vectorized ds_load_b128).
// Pass 2 over j (chunks of 32): build normalized P tile directly in the WMMA
// 16-bit A layout, stream the attention matrix out once with non-temporal
// b128 stores (604 MB, the roofline term; NT keeps hpT resident in L2), and
// accumulate P @ h_prime with v_wmma_f32_16x16x32_bf16 (fp32 accumulators).
// ---------------------------------------------------------------------------
__global__ void __launch_bounds__(256)
gat_attn_kernel(const float* __restrict__ attn_src,
                const float* __restrict__ attn_dst,
                const float* __restrict__ dmaxp,
                const __bf16* __restrict__ hpT,
                const float* __restrict__ bias,
                float* __restrict__ out_feat,
                float* __restrict__ out_attn) {
    __shared__ float ds[NN];                       // 24 KB: attn_dst for this head
    const int head = blockIdx.y;
    const int tid  = threadIdx.x;
    for (int j = tid; j < NN; j += 256)
        ds[j] = attn_dst[(size_t)head * NN + j];
    __syncthreads();

    const int lane = tid & 31;
    const int wv   = tid >> 5;
    const int r0   = (blockIdx.x * 8 + wv) * 16;   // this wave's 16-row tile
    const int M    = lane & 15;                    // A-matrix row for this lane
    const int row  = r0 + M;

    const float s = attn_src[(size_t)head * NN + row];
    const float m = lrelu(s + dmaxp[head]);        // exact row max

    // ---- pass 1: softmax denominator (lanes 0-15 / 16-31 split the range) ----
    float den = 0.f;
    const v4f* ds4 = (const v4f*)ds;
    for (int j4 = (lane >> 4); j4 < NN / 4; j4 += 2) {
        v4f v = ds4[j4];                           // ds_load_b128
        den += __expf(lrelu(s + v.x) - m);
        den += __expf(lrelu(s + v.y) - m);
        den += __expf(lrelu(s + v.z) - m);
        den += __expf(lrelu(s + v.w) - m);
    }
    den += __shfl_xor(den, 16, 32);
    const float rden = 1.f / den;

    // ---- pass 2: normalized P tiles -> attn out + WMMA accumulate ----
    const int c0 = (lane < 16) ? 0 : 8;            // A K-block base (16-bit A layout)
    const int kb = (lane < 16) ? 0 : 8;            // B K-block base (matches A: kb..kb+7, kb+16..kb+23)
    const int ob = lane & 15;                      // B column (output feature)
    const __bf16* b0p = hpT + ((size_t)head * F_OUT + ob) * NN;        // N half 0
    const __bf16* b1p = hpT + ((size_t)head * F_OUT + ob + 16) * NN;   // N half 1
    float* arow = out_attn + ((size_t)head * NN + row) * NN;

    v8f acc0 = {};
    v8f acc1 = {};
    for (int j0 = 0; j0 < NN; j0 += 32) {
        float p[16];
#pragma unroll
        for (int t = 0; t < 8; ++t)
            p[t]     = __expf(lrelu(s + ds[j0 + c0 + t])      - m) * rden;
#pragma unroll
        for (int t = 0; t < 8; ++t)
            p[8 + t] = __expf(lrelu(s + ds[j0 + c0 + 16 + t]) - m) * rden;

        // stream normalized attention tile: non-temporal b128 stores
        v4f q0 = {p[0],  p[1],  p[2],  p[3]};
        v4f q1 = {p[4],  p[5],  p[6],  p[7]};
        v4f q2 = {p[8],  p[9],  p[10], p[11]};
        v4f q3 = {p[12], p[13], p[14], p[15]};
        __builtin_nontemporal_store(q0, (v4f*)(arow + j0 + c0));
        __builtin_nontemporal_store(q1, (v4f*)(arow + j0 + c0 + 4));
        __builtin_nontemporal_store(q2, (v4f*)(arow + j0 + c0 + 16));
        __builtin_nontemporal_store(q3, (v4f*)(arow + j0 + c0 + 20));

        // A operand: 16x32 bf16, lane M=lane&15, K = {c0..c0+7, c0+16..c0+23}
        v16bf a;
#pragma unroll
        for (int t = 0; t < 16; ++t) a[t] = (__bf16)p[t];

        // B operand: 32x16 bf16, two 16B chunks per lane (K = kb..kb+7, kb+16..kb+23) from transposed h_prime
        typedef __attribute__((ext_vector_type(8))) __bf16 v8bf_;
        v16bf b0 = __builtin_shufflevector(*(const v8bf_*)(b0p + j0 + kb), *(const v8bf_*)(b0p + j0 + kb + 16), 0,1,2,3,4,5,6,7,8,9,10,11,12,13,14,15);
        v16bf b1 = __builtin_shufflevector(*(const v8bf_*)(b1p + j0 + kb), *(const v8bf_*)(b1p + j0 + kb + 16), 0,1,2,3,4,5,6,7,8,9,10,11,12,13,14,15);
        __builtin_prefetch(b0p + j0 + kb + 32, 0, 3);   // global_prefetch next K-chunk
        __builtin_prefetch(b1p + j0 + kb + 32, 0, 3);

        acc0 = __builtin_amdgcn_wmma_f32_16x16x32_bf16(false, a, false, b0,
                                                       (short)0, acc0, false, false);
        acc1 = __builtin_amdgcn_wmma_f32_16x16x32_bf16(false, a, false, b1,
                                                       (short)0, acc1, false, false);
    }

    // ---- store output features (C/D layout: VGPR r -> M=r (+8 for hi lanes)) ----
    const int mo = (lane < 16) ? 0 : 8;
    const float bia0 = bias[ob];
    const float bia1 = bias[ob + 16];
#pragma unroll
    for (int r = 0; r < 8; ++r) {
        const int orow = r0 + r + mo;
        out_feat[((size_t)head * NN + orow) * F_OUT + ob]      = acc0[r] + bia0;
        out_feat[((size_t)head * NN + orow) * F_OUT + ob + 16] = acc1[r] + bia1;
    }
}

// ---------------------------------------------------------------------------
extern "C" void kernel_launch(void* const* d_in, const int* in_sizes, int n_in,
                              void* d_out, int out_size, void* d_ws, size_t ws_size,
                              hipStream_t stream) {
    const float* h     = (const float*)d_in[0];
    const float* w     = (const float*)d_in[1];
    const float* a_src = (const float*)d_in[2];
    const float* a_dst = (const float*)d_in[3];
    const float* bias  = (const float*)d_in[4];

    float* out_feat = (float*)d_out;                                    // [4,6144,32]
    float* out_attn = out_feat + (size_t)N_HEAD * NN * F_OUT;           // [4,6144,6144]

    char*  ws  = (char*)d_ws;
    __bf16* hpT = (__bf16*)ws;                                          // [4][32][6144] bf16
    size_t off = (size_t)N_HEAD * F_OUT * NN * sizeof(__bf16);          // 1.5 MB
    float* attn_src = (float*)(ws + off); off += (size_t)N_HEAD * NN * sizeof(float);
    float* attn_dst = (float*)(ws + off); off += (size_t)N_HEAD * NN * sizeof(float);
    float* dmax     = (float*)(ws + off);

    dim3 g1(NN / 8, N_HEAD);
    gat_proj_kernel<<<g1, 256, 0, stream>>>(h, w, a_src, a_dst, hpT, attn_src, attn_dst);
    gat_dmax_kernel<<<N_HEAD, 256, 0, stream>>>(attn_dst, dmax);
    dim3 g2(NN / 128, N_HEAD);
    gat_attn_kernel<<<g2, 256, 0, stream>>>(attn_src, attn_dst, dmax, hpT,
                                            bias, out_feat, out_attn);
}
